// AutoRegressiveAdaptiveSpectralConv2d_56650618635113
// MI455X (gfx1250) — compile-verified, matrix operand order checked by reading, not run on hardware
//
#include <hip/hip_runtime.h>
#include <hip/hip_bf16.h>

// Problem constants (match reference)
#define B_  4
#define T_  10
#define U_  64
#define XR_ 128
#define YR_ 128
#define XM_ 32
#define YM_ 32

typedef __attribute__((ext_vector_type(16))) _Float16 v16h;
typedef __attribute__((ext_vector_type(8)))  _Float16 v8h;
typedef __attribute__((ext_vector_type(4)))  _Float16 v4h;
typedef __attribute__((ext_vector_type(8)))  float    v8f;
typedef __attribute__((ext_vector_type(4)))  float    v4f;

typedef unsigned int u32x4 __attribute__((ext_vector_type(4)));
typedef int          i32x8 __attribute__((ext_vector_type(8)));
typedef int          i32x4 __attribute__((ext_vector_type(4)));

#if __has_builtin(__builtin_amdgcn_tensor_load_to_lds) && \
    __has_builtin(__builtin_amdgcn_s_wait_tensorcnt)
#define USE_TDM 1
#else
#define USE_TDM 0
#endif

__device__ __forceinline__ v8f wmma16(v16h a, v16h b, v8f c) {
  return __builtin_amdgcn_wmma_f32_16x16x32_f16(false, a, false, b, (short)0, c,
                                                false, false);
}

// Unified operand load (A: src[m][k] row-major; B: srcT[n][k] row-major).
// lane<16: row=lane,   K in {0..7,16..23}; lane>=16: row=lane-16, K in {8..15,24..31}
// Both 8-element K runs are contiguous -> two 16B vector loads.
__device__ __forceinline__ v16h load_op(const _Float16* src, int ld, int lane) {
  int row = lane & 15;
  int kb  = (lane < 16) ? 0 : 8;
  const _Float16* r = src + row * ld + kb;
  v8h lo = *(const v8h*)(r);
  v8h hi = *(const v8h*)(r + 16);
  return __builtin_shufflevector(lo, hi, 0, 1, 2, 3, 4, 5, 6, 7,
                                 8, 9, 10, 11, 12, 13, 14, 15);
}

// C/D tile element v -> row mb+v, col = lane&15 (16x16 f32)
// Row-major half store (strided u16 per element)
__device__ __forceinline__ void store_tile_h(v8f acc, _Float16* dst, int ld, int lane) {
  int col = lane & 15;
  int mb  = (lane < 16) ? 0 : 8;
#pragma unroll
  for (int v = 0; v < 8; ++v) dst[(mb + v) * ld + col] = (_Float16)acc[v];
}

// Transposed half store: dstT[n][m]; per-lane M-run is contiguous -> one 16B store
__device__ __forceinline__ void store_tile_hT(v8f acc, _Float16* dstT, int ldT, int lane) {
  int n  = lane & 15;
  int mb = (lane < 16) ? 0 : 8;
  v8h h;
#pragma unroll
  for (int v = 0; v < 8; ++v) h[v] = (_Float16)acc[v];
  *(v8h*)(dstT + n * ldT + mb) = h;
}

// LDS offset of a __shared__ object: generic LDS address carries offset in [31:0]
__device__ __forceinline__ unsigned int lds_off_u32(const void* p) {
  return (unsigned int)(unsigned long long)(uintptr_t)p;
}

#if USE_TDM
// TDM: contiguous 1-D tile of n f16 elements, global -> LDS (6-arg toolchain form)
__device__ __forceinline__ void tdm_load_f16_1d(unsigned int lds_off,
                                                const void* gaddr, unsigned int n) {
  unsigned long long ga = (unsigned long long)(uintptr_t)gaddr;
  u32x4 g0;
  g0[0] = 1u;                                            // count=1, user mode
  g0[1] = lds_off;                                       // lds_addr
  g0[2] = (unsigned int)(ga & 0xffffffffu);              // global_addr[31:0]
  g0[3] = (unsigned int)((ga >> 32) & 0x01ffffffu) | (2u << 30); // [56:32] | type=2
  i32x8 g1;
  g1[0] = (int)(1u << 16);                               // data_size=1 (2 bytes)
  g1[1] = (int)((n & 0xffffu) << 16);                    // tensor_dim0[15:0]
  g1[2] = (int)(((n >> 16) & 0xffffu) | (1u << 16));     // dim0[31:16] | tensor_dim1=1
  g1[3] = (int)((n & 0xffffu) << 16);                    // tile_dim0 = n
  g1[4] = 1;                                             // tile_dim1 = 1
  g1[5] = (int)n;                                        // tensor_dim0_stride[31:0]
  g1[6] = (int)((n & 0xffffu) << 16);                    // tensor_dim1_stride[15:0]
  g1[7] = 0;
  i32x4 z4 = {0, 0, 0, 0};                               // D# groups 2/3 unused (<=2D)
  i32x8 z8 = {0, 0, 0, 0, 0, 0, 0, 0};
  __builtin_amdgcn_tensor_load_to_lds(g0, g1, z4, z4, z8, 0);
}
#endif

// ---------------------------------------------------------------------------
// Twiddle tables (f16, 10 tables x 4096):
// 0 FyCt[ky][y] ld128:  cos(2pi y ky/128)            (fwd stage1 B^T, e^-i)
// 1 FySt[ky][y] ld128: -sin
// 2 FxC [kx][x] ld128:  cos(2pi kx x/128)            (fwd stage2 A)
// 3 FxSp[kx][x] ld128: +sin
// 4 FxSn[kx][x] ld128: -sin
// 5 EiC [x][kx] ld32 :  cos(2pi x kx/128)/128        (inv stage A, e^+i)
// 6 EiS [x][kx] ld32 : +sin/128
// 7 EiSn[x][kx] ld32 : -sin/128
// 8 GTc [y][ky] ld32 :  w(ky) cos(2pi ky y/128)/128  (inv stage B A-side)
// 9 GTs [y][ky] ld32 : -w(ky) sin(...)/128,  w(0)=1, w(k>0)=2
// ---------------------------------------------------------------------------
__global__ void spectral_init_tables(_Float16* __restrict__ tab) {
  int idx = blockIdx.x * blockDim.x + threadIdx.x;
  if (idx >= 10 * 4096) return;
  int tb  = idx >> 12;
  int pos = idx & 4095;
  const float w = 0.049087385212340517f; // 2*pi/128
  float val = 0.0f;
  if (tb <= 1) {                       // [ky][y], ld 128
    int ky = pos >> 7, y = pos & 127;
    float th = (float)((y * ky) & 127) * w;
    val = (tb == 0) ? cosf(th) : -sinf(th);
  } else if (tb <= 4) {                // [kx][x], ld 128
    int kx = pos >> 7, x = pos & 127;
    float th = (float)((kx * x) & 127) * w;
    val = (tb == 2) ? cosf(th) : ((tb == 3) ? sinf(th) : -sinf(th));
  } else if (tb <= 7) {                // [x][kx], ld 32, scaled 1/128
    int x = pos >> 5, kx = pos & 31;
    float th = (float)((x * kx) & 127) * w;
    float c = cosf(th) * (1.0f / 128.0f), s = sinf(th) * (1.0f / 128.0f);
    val = (tb == 5) ? c : ((tb == 6) ? s : -s);
  } else {                             // [y][ky], ld 32, Hermitian weights /128
    int y = pos >> 5, ky = pos & 31;
    float th = (float)((ky * y) & 127) * w;
    float wt = (ky == 0) ? 1.0f : 2.0f;
    val = (tb == 8) ? wt * cosf(th) * (1.0f / 128.0f)
                    : -wt * sinf(th) * (1.0f / 128.0f);
  }
  tab[idx] = (_Float16)val;
}

// ---------------------------------------------------------------------------
// Forward: per (b,t,u) slice: f = trunc-FFT2(X)[:32,:32] * Ws_w[u]
// stage1: p(128x32,cplx) = X(128x128) * Fy(128x32); stored transposed pT[ky][x]
// stage2: f_r = FxC*p_r + FxSp*p_i ; f_i = FxC*p_i + FxSn*p_r
// fbuf layout: [t][kx*32+ky][i][b]
// ---------------------------------------------------------------------------
__global__ __launch_bounds__(256) void spectral_forward(
    const float* __restrict__ X, const float* __restrict__ Ws,
    const _Float16* __restrict__ tab,
    _Float16* __restrict__ fr, _Float16* __restrict__ fi) {
  __shared__ __attribute__((aligned(16))) _Float16 Xh[XR_ * YR_];   // 32 KB
  __shared__ __attribute__((aligned(16))) _Float16 pT[2][YM_ * XR_];// 16 KB [ky][x]

  int slice = blockIdx.x;                   // ((b*T + t)*U + u)
  int b = slice / (T_ * U_);
  int t = (slice / U_) % T_;
  int u = slice % U_;
  int tid = threadIdx.x, wave = tid >> 5, lane = tid & 31;

  const float* Xs = X + (size_t)slice * (XR_ * YR_);
  if (slice + 1 < B_ * T_ * U_)
    __builtin_prefetch(X + (size_t)(slice + 1) * (XR_ * YR_), 0, 1);

  for (int j = tid; j < (XR_ * YR_) / 4; j += 256) {
    v4f x = ((const v4f*)Xs)[j];
    ((v4h*)Xh)[j] = __builtin_convertvector(x, v4h);
  }
  __syncthreads();

  const _Float16* FyCt = tab;              // [ky][y] ld128
  const _Float16* FySt = tab + 4096;

  // stage1: 2 planes x 8 mtiles(x) x 2 ntiles(ky) = 32 jobs, 8 waves x 4
#pragma unroll
  for (int jj = 0; jj < 4; ++jj) {
    int job = wave + 8 * jj;
    int plane = job >> 4, mt = (job >> 1) & 7, nt = job & 1;
    const _Float16* Bt = plane ? FySt : FyCt;
    v8f acc = {};
#pragma unroll
    for (int kk = 0; kk < 4; ++kk) {
      v16h a  = load_op(Xh + mt * 16 * YR_ + kk * 32, YR_, lane);
      v16h bb = load_op(Bt + nt * 16 * YR_ + kk * 32, YR_, lane);
      acc = wmma16(a, bb, acc);
    }
    store_tile_hT(acc, &pT[plane][0] + nt * 16 * XR_ + mt * 16, XR_, lane);
  }
  __syncthreads();

  const _Float16* FxC  = tab + 2 * 4096;
  const _Float16* FxSp = tab + 3 * 4096;
  const _Float16* FxSn = tab + 4 * 4096;

  // stage2: 8 jobs (2 planes x 2 mtiles(kx) x 2 ntiles(ky)), one per wave
  int plane = wave >> 2, mt = (wave >> 1) & 1, nt = wave & 1;
  const _Float16* A2 = plane ? FxSn : FxSp;
  const _Float16* B1 = plane ? &pT[1][0] : &pT[0][0];
  const _Float16* B2 = plane ? &pT[0][0] : &pT[1][0];
  v8f acc = {};
#pragma unroll
  for (int kk = 0; kk < 4; ++kk)
    acc = wmma16(load_op(FxC + mt * 16 * XR_ + kk * 32, XR_, lane),
                 load_op(B1 + nt * 16 * XR_ + kk * 32, XR_, lane), acc);
#pragma unroll
  for (int kk = 0; kk < 4; ++kk)
    acc = wmma16(load_op(A2 + mt * 16 * XR_ + kk * 32, XR_, lane),
                 load_op(B2 + nt * 16 * XR_ + kk * 32, XR_, lane), acc);

  int col = lane & 15, mb = (lane < 16) ? 0 : 8;
  int ky = nt * 16 + col;
  _Float16* dstp = plane ? fi : fr;
#pragma unroll
  for (int v = 0; v < 8; ++v) {
    int kx = mt * 16 + mb + v;
    float s = Ws[(size_t)u * (XM_ * YM_) + kx * YM_ + ky];
    size_t didx = ((size_t)t * (XM_ * YM_) + (size_t)(kx * YM_ + ky)) * (U_ * B_)
                  + (size_t)u * B_ + b;
    dstp[didx] = (_Float16)(acc[v] * s);
  }
}

// ---------------------------------------------------------------------------
// Channel mixing per (t,kx,ky): g(o,b) = W^T(o,i) f(i,b), complex
//   g_r = Wr^T f_r + Wi^T (-f_i) ;  g_i = Wi^T f_r + Wr^T f_i
// gbuf layout (pre-transposed for inverse): [t][o][b][ky*32+kx]
// ---------------------------------------------------------------------------
__global__ __launch_bounds__(256) void spectral_mix(
    const float* __restrict__ R_wr, const float* __restrict__ R_wi,
    const _Float16* __restrict__ fr, const _Float16* __restrict__ fi,
    _Float16* __restrict__ gr, _Float16* __restrict__ gi) {
  __shared__ __attribute__((aligned(16))) _Float16 WrT[U_ * U_];   // [o][i] 8KB
  __shared__ __attribute__((aligned(16))) _Float16 WiT[U_ * U_];
  __shared__ __attribute__((aligned(16))) _Float16 fBrT[16 * U_];  // [b][i] 2KB
  __shared__ __attribute__((aligned(16))) _Float16 fBiT[16 * U_];
  __shared__ __attribute__((aligned(16))) _Float16 fBinT[16 * U_]; // -f_i

  int mode = blockIdx.x;               // t*1024 + xy, xy = kx*32+ky
  int t = mode / (XM_ * YM_);
  int xy = mode % (XM_ * YM_);
  int tid = threadIdx.x, wave = tid >> 5, lane = tid & 31;

  for (int idx = tid; idx < U_ * U_; idx += 256) {
    int i = idx >> 6, o = idx & 63;
    size_t base = (((size_t)t * U_ + i) * U_ + o) * (XM_ * YM_) + xy;
    WrT[o * U_ + i] = (_Float16)R_wr[base];
    WiT[o * U_ + i] = (_Float16)R_wi[base];
  }
  for (int idx = tid; idx < 16 * U_; idx += 256) {
    int c = idx >> 6, i = idx & 63;
    _Float16 vr = (_Float16)0.0f, vi = (_Float16)0.0f;
    if (c < B_) {
      size_t base = ((size_t)t * (XM_ * YM_) + xy) * (U_ * B_) + (size_t)i * B_ + c;
      vr = fr[base];
      vi = fi[base];
    }
    fBrT[idx] = vr;
    fBiT[idx] = vi;
    fBinT[idx] = (_Float16)(-(float)vi);
  }
  __syncthreads();

  // 8 jobs: plane (g_r/g_i) x 4 M-tiles(o), one per wave
  int plane = wave >> 2, mt = wave & 3;
  const _Float16* A1 = plane ? WiT : WrT;    // pairs with f_r
  const _Float16* A2 = plane ? WrT : WiT;    // pairs with f_i / -f_i
  const _Float16* Bf2 = plane ? fBiT : fBinT;
  v8f acc = {};
#pragma unroll
  for (int kk = 0; kk < 2; ++kk)
    acc = wmma16(load_op(A1 + mt * 16 * U_ + kk * 32, U_, lane),
                 load_op(fBrT + kk * 32, U_, lane), acc);
#pragma unroll
  for (int kk = 0; kk < 2; ++kk)
    acc = wmma16(load_op(A2 + mt * 16 * U_ + kk * 32, U_, lane),
                 load_op(Bf2 + kk * 32, U_, lane), acc);

  int col = lane & 15, mb = (lane < 16) ? 0 : 8;
  int pos = (xy & 31) * 32 + (xy >> 5);      // transposed: ky*32+kx
  if (col < B_) {
    _Float16* dstp = plane ? gi : gr;
#pragma unroll
    for (int v = 0; v < 8; ++v) {
      int o = mt * 16 + mb + v;
      size_t didx = ((((size_t)t * U_ + o) * B_) + col) * (XM_ * YM_) + pos;
      dstp[didx] = (_Float16)acc[v];
    }
  }
}

// ---------------------------------------------------------------------------
// Inverse: per (b,t,o) slice: zero-padded irfft2 of g(32x32 complex)
// stage A: h(128x32,cplx) = Ei(128x32) * g(32x32)  (B^T = gT[ky][kx], via TDM)
//   h_r = EiC g_r + EiSn g_i ;  h_i = EiC g_i + EiS g_r    -> h[x][ky] in LDS
// stage B (transposed): out^T(y,x) = GTc*h_r^T + GTs*h_i^T
//   A = GT[y][ky], B^T = h[x][ky]; contiguous v8f global stores
// ---------------------------------------------------------------------------
__global__ __launch_bounds__(256) void spectral_inverse(
    const _Float16* __restrict__ gr, const _Float16* __restrict__ gi,
    const _Float16* __restrict__ tab, float* __restrict__ out) {
  __shared__ __attribute__((aligned(16))) _Float16 sgrT[YM_ * XM_]; // [ky][kx] 2KB
  __shared__ __attribute__((aligned(16))) _Float16 sgiT[YM_ * XM_];
  __shared__ __attribute__((aligned(16))) _Float16 hr[XR_ * YM_];   // [x][ky] 8KB
  __shared__ __attribute__((aligned(16))) _Float16 hi[XR_ * YM_];

  int slice = blockIdx.x;              // ((b*T + t)*U + o), output-contiguous
  int b = slice / (T_ * U_);
  int t = (slice / U_) % T_;
  int o = slice % U_;
  int tid = threadIdx.x, wave = tid >> 5, lane = tid & 31;

  size_t gbase = ((((size_t)t * U_ + o) * B_) + b) * (XM_ * YM_);
#if USE_TDM
  if (wave == 0) {
    tdm_load_f16_1d(lds_off_u32(&sgrT[0]), gr + gbase, XM_ * YM_);
    tdm_load_f16_1d(lds_off_u32(&sgiT[0]), gi + gbase, XM_ * YM_);
    __builtin_amdgcn_s_wait_tensorcnt(0);
  }
#else
  for (int idx = tid; idx < (XM_ * YM_) / 8; idx += 256) {
    ((uint4*)sgrT)[idx] = ((const uint4*)(gr + gbase))[idx];
    ((uint4*)sgiT)[idx] = ((const uint4*)(gi + gbase))[idx];
  }
#endif
  __syncthreads();

  const _Float16* EiC  = tab + 5 * 4096;
  const _Float16* EiS  = tab + 6 * 4096;
  const _Float16* EiSn = tab + 7 * 4096;

  // stage A: 2 planes x 8 mtiles(x) x 2 ntiles(ky) = 32 jobs, 8 waves x 4
#pragma unroll
  for (int jj = 0; jj < 4; ++jj) {
    int job = wave + 8 * jj;
    int plane = job >> 4, mt = (job >> 1) & 7, nt = job & 1;
    const _Float16* Ay = plane ? EiS : EiSn;
    const _Float16* Bx = plane ? sgiT : sgrT;
    const _Float16* By = plane ? sgrT : sgiT;
    v8f acc = {};
    acc = wmma16(load_op(EiC + mt * 16 * XM_, XM_, lane),
                 load_op(Bx + nt * 16 * XM_, XM_, lane), acc);
    acc = wmma16(load_op(Ay + mt * 16 * XM_, XM_, lane),
                 load_op(By + nt * 16 * XM_, XM_, lane), acc);
    store_tile_h(acc, (plane ? hi : hr) + mt * 16 * YM_ + nt * 16, YM_, lane);
  }
  __syncthreads();

  const _Float16* GTc = tab + 8 * 4096;    // [y][ky] ld32
  const _Float16* GTs = tab + 9 * 4096;
  float* outs = out + (size_t)slice * (XR_ * YR_);

  // stage B: 8 mtiles(y) x 8 ntiles(x) = 64 jobs, 8 waves x 8
#pragma unroll
  for (int jj = 0; jj < 8; ++jj) {
    int job = wave + 8 * jj;
    int mt = job >> 3, nt = job & 7;
    v8f acc = {};
    acc = wmma16(load_op(GTc + mt * 16 * YM_, YM_, lane),
                 load_op(hr + nt * 16 * YM_, YM_, lane), acc);
    acc = wmma16(load_op(GTs + mt * 16 * YM_, YM_, lane),
                 load_op(hi + nt * 16 * YM_, YM_, lane), acc);
    // element v: y = mt*16+mb+v, x = nt*16+col -> contiguous 8 floats in y
    int col = lane & 15, mb = (lane < 16) ? 0 : 8;
    *(v8f*)(outs + (size_t)(nt * 16 + col) * YR_ + mt * 16 + mb) = acc;
  }
}

extern "C" void kernel_launch(void* const* d_in, const int* in_sizes, int n_in,
                              void* d_out, int out_size, void* d_ws, size_t ws_size,
                              hipStream_t stream) {
  const float* X    = (const float*)d_in[0];
  const float* R_wr = (const float*)d_in[1];
  const float* R_wi = (const float*)d_in[2];
  const float* Ws   = (const float*)d_in[3];
  float* out = (float*)d_out;

  char* ws = (char*)d_ws;
  auto align256 = [](size_t x) { return (x + 255) & ~(size_t)255; };
  size_t off = 0;
  _Float16* tab = (_Float16*)(ws + off);
  off = align256(off + (size_t)10 * 4096 * sizeof(_Float16));
  const size_t FC = (size_t)T_ * XM_ * YM_ * U_ * B_;   // 2,621,440 halves
  _Float16* fr = (_Float16*)(ws + off); off = align256(off + FC * sizeof(_Float16));
  _Float16* fi = (_Float16*)(ws + off); off = align256(off + FC * sizeof(_Float16));
  _Float16* gr = (_Float16*)(ws + off); off = align256(off + FC * sizeof(_Float16));
  _Float16* gi = (_Float16*)(ws + off); off = align256(off + FC * sizeof(_Float16));
  (void)ws_size; (void)in_sizes; (void)n_in; (void)out_size;

  spectral_init_tables<<<(10 * 4096 + 255) / 256, 256, 0, stream>>>(tab);
  spectral_forward<<<B_ * T_ * U_, 256, 0, stream>>>(X, Ws, tab, fr, fi);
  spectral_mix<<<T_ * XM_ * YM_, 256, 0, stream>>>(R_wr, R_wi, fr, fi, gr, gi);
  spectral_inverse<<<B_ * T_ * U_, 256, 0, stream>>>(gr, gi, tab, out);
}
